// CRF_29076928594275
// MI455X (gfx1250) — matrix operand order repaired and checked by reading, not compiled, not run on hardware
//
#include <hip/hip_runtime.h>
#include <stdint.h>

#define BB   2048
#define SS   512
#define LL   32
#define TILE 16

#define LOG2E 1.4426950408889634f
#define LN2   0.6931471805599453f

typedef __attribute__((ext_vector_type(16))) _Float16 v16h;
typedef __attribute__((ext_vector_type(8)))  float    v8f;

#if __has_builtin(__builtin_amdgcn_exp2f) && __has_builtin(__builtin_amdgcn_logf)
// raw v_exp_f32 / v_log_f32 (base-2); valid: args never denormal/inf in our ranges
#define FAST_EXP2(x) __builtin_amdgcn_exp2f(x)
#define FAST_LOG2(x) __builtin_amdgcn_logf(x)
#else
#define FAST_EXP2(x) exp2f(x)
#define FAST_LOG2(x) log2f(x)
#endif

#if __has_builtin(__builtin_amdgcn_global_load_async_to_lds_b128) && \
    __has_builtin(__builtin_amdgcn_s_wait_asynccnt)
#define HAVE_ASYNC 1
// per compiler diagnostic: params are pointers to 16B int vectors (global / LDS)
typedef int v4i_t __attribute__((ext_vector_type(4)));
typedef __attribute__((address_space(1))) v4i_t gv4i_t;
typedef __attribute__((address_space(3))) v4i_t lv4i_t;
#endif

// A-matrix (16x32 f16) per-lane K index for vector element e (0..15)
__device__ __forceinline__ int kmapA(int e, int hi) {
    return (e < 8) ? (e + 8 * hi) : (e + 8 + 8 * hi);
}
// B-matrix (32x16 f16) per-lane K index for vector element e (0..15)
__device__ __forceinline__ int kmapB(int e, int hi) {
    return (e < 8) ? (e + 8 * hi) : (e + 8 + 8 * hi);
}

// ---------------------------------------------------------------------------
// Forward algorithm: one wave32 per 16-batch tile, recurrence via WMMA.
// score'[r][j] = rowmax[r] + log( sum_i exp(score[r][i]-rowmax[r]) * E[i][j] ) + em[r][t][j]
// ---------------------------------------------------------------------------
__global__ __launch_bounds__(32) void crf_forward(
    const float* __restrict__ em,      // [BB][SS][LL]
    const uint8_t* __restrict__ mask,  // [BB][SS]
    const float* __restrict__ trans,   // [LL][LL]
    const float* __restrict__ start,   // [LL]
    const float* __restrict__ endt,    // [LL]
    float* __restrict__ fwd_out)       // [BB]
{
    // scores: 144B row stride keeps 16B alignment for b128 row reads, no bank alias
    __shared__ __attribute__((aligned(16))) float sc[TILE][36];
    __shared__ __attribute__((aligned(16))) float rmbuf[TILE];  // per-row max
    __shared__ __attribute__((aligned(16))) float mkbuf[TILE];  // per-row mask
#ifdef HAVE_ASYNC
    // double-buffered emission tiles (async b128 dst)
    __shared__ __attribute__((aligned(16))) float embuf[2][TILE][36];
#endif

    const int lane  = threadIdx.x;
    const int col   = lane & 15;       // matrix column / row-id this lane owns
    const int hi    = lane >> 4;       // which half of the wave
    const int rbase = hi * 8;          // D-matrix rows held by this lane
    const int b0    = blockIdx.x * TILE;

    // Constant B operands: E = exp(trans), split into label columns 0..15 / 16..31.
    v16h Bm0, Bm1;
#pragma unroll
    for (int e = 0; e < 16; ++e) {
        int k = kmapB(e, hi);
        Bm0[e] = (_Float16)__expf(trans[k * LL + col]);
        Bm1[e] = (_Float16)__expf(trans[k * LL + col + 16]);
    }

    // score0 = start + emissions[:,0]; lane (col,hi) writes row=col, labels hi*16..+15
    {
        const float* emr = em + (size_t)(b0 + col) * SS * LL;
#pragma unroll
        for (int j = 0; j < 16; ++j) {
            int jj = hi * 16 + j;
            sc[col][jj] = start[jj] + emr[jj];
        }
    }
    __syncthreads();

#ifdef HAVE_ASYNC
    // Async DMA of one 16x32 f32 emission tile: lane l chunk q covers
    // row = q*4 + (l>>3), 16B at column (l&7)*4 -> 4 x b128 per wave-lane = 2KB.
    const int arow = lane >> 3;        // 0..3
    const int aj0  = (lane & 7) * 4;   // 0,4,...,28
#define ISSUE_TILE(t_, buf_)                                                    \
    {                                                                           \
        _Pragma("unroll")                                                       \
        for (int q = 0; q < 4; ++q) {                                           \
            int r_ = q * 4 + arow;                                              \
            const float* g_ = em + ((size_t)(b0 + r_) * SS + (t_)) * LL + aj0;  \
            __builtin_amdgcn_global_load_async_to_lds_b128(                     \
                (gv4i_t*)g_, (lv4i_t*)&embuf[buf_][r_][aj0], 0, 0);             \
        }                                                                       \
    }
    ISSUE_TILE(1, 1);
#else
    // Fallback: software-pipelined register prefetch (coalesced 64B rows)
    float pe0[8], pe1[8];
#pragma unroll
    for (int v = 0; v < 8; ++v) {
        const float* p = em + ((size_t)(b0 + rbase + v) * SS + 1) * LL;
        pe0[v] = p[col];
        pe1[v] = p[col + 16];
    }
#endif

    for (int t = 1; t < SS; ++t) {
#ifdef HAVE_ASYNC
        const int cur = t & 1;
        if (t + 1 < SS) {
            ISSUE_TILE(t + 1, cur ^ 1);
            // async loads retire in order: <=4 outstanding => tile t resident
            __builtin_amdgcn_s_wait_asynccnt(4);
        } else {
            __builtin_amdgcn_s_wait_asynccnt(0);
        }
#else
        float e0[8], e1[8];
#pragma unroll
        for (int v = 0; v < 8; ++v) { e0[v] = pe0[v]; e1[v] = pe1[v]; }
        if (t + 1 < SS) {
#pragma unroll
            for (int v = 0; v < 8; ++v) {
                const float* p = em + ((size_t)(b0 + rbase + v) * SS + (t + 1)) * LL;
                pe0[v] = p[col];
                pe1[v] = p[col + 16];
            }
        }
#endif
        // mask byte for row 'col' at step t (lanes 16..31 hold duplicates)
        float mval = (float)mask[(size_t)(b0 + col) * SS + t];

        // full current score row via 8 x ds_load_b128 -> local max (no cross-lane)
        float row[32];
        {
            const float4* rp = (const float4*)(&sc[col][0]);
#pragma unroll
            for (int q = 0; q < 8; ++q) {
                float4 rv = rp[q];
                row[q * 4 + 0] = rv.x; row[q * 4 + 1] = rv.y;
                row[q * 4 + 2] = rv.z; row[q * 4 + 3] = rv.w;
            }
        }
        float m = row[0];
#pragma unroll
        for (int j = 1; j < 32; ++j) m = fmaxf(m, row[j]);

        // stage per-row scalars for the epilogue (upper half writes duplicates)
        rmbuf[col] = m;
        mkbuf[col] = mval;

        // A = exp2((score - rowmax)*log2e) packed into WMMA A layout
        float msc = m * LOG2E;
        v16h A;
#pragma unroll
        for (int e = 0; e < 16; ++e) {
            A[e] = (_Float16)FAST_EXP2(__builtin_fmaf(row[kmapA(e, hi)], LOG2E, -msc));
        }

        v8f z = {0.f, 0.f, 0.f, 0.f, 0.f, 0.f, 0.f, 0.f};
        v8f D0 = __builtin_amdgcn_wmma_f32_16x16x32_f16(false, A, false, Bm0,
                                                        (short)0, z, false, false);
        v8f D1 = __builtin_amdgcn_wmma_f32_16x16x32_f16(false, A, false, Bm1,
                                                        (short)0, z, false, false);

        __syncthreads();   // everyone done reading old scores before overwrite

        // batched epilogue loads (broadcast b128 + merged 2addr pairs)
        float4 rma = *(const float4*)(&rmbuf[rbase]);
        float4 rmb = *(const float4*)(&rmbuf[rbase + 4]);
        float4 mka = *(const float4*)(&mkbuf[rbase]);
        float4 mkb = *(const float4*)(&mkbuf[rbase + 4]);
        float mrv[8] = {rma.x, rma.y, rma.z, rma.w, rmb.x, rmb.y, rmb.z, rmb.w};
        float mkv[8] = {mka.x, mka.y, mka.z, mka.w, mkb.x, mkb.y, mkb.z, mkb.w};

        float e0v[8], e1v[8], o0[8], o1[8];
#pragma unroll
        for (int v = 0; v < 8; ++v) {
            int r = rbase + v;
#ifdef HAVE_ASYNC
            e0v[v] = embuf[cur][r][col];
            e1v[v] = embuf[cur][r][col + 16];
#else
            e0v[v] = e0[v];
            e1v[v] = e1[v];
#endif
            o0[v] = sc[r][col];        // old scores for mask select (branch-free)
            o1[v] = sc[r][col + 16];
        }
#pragma unroll
        for (int v = 0; v < 8; ++v) {
            int r = rbase + v;
            // n = rowmax + ln2*log2(sum) + em ; sums in [0.6,48] -> raw v_log ok
            float n0 = __builtin_fmaf(LN2, FAST_LOG2(D0[v]), mrv[v]) + e0v[v];
            float n1 = __builtin_fmaf(LN2, FAST_LOG2(D1[v]), mrv[v]) + e1v[v];
            bool keep = (mkv[v] != 0.f);      // jnp.where(mask, nxt, score)
            sc[r][col]      = keep ? n0 : o0[v];
            sc[r][col + 16] = keep ? n1 : o1[v];
        }
        __syncthreads();
    }

    // fwd[b] = logsumexp_j(score[b][j] + end[j])
    float vals[32];
    {
        const float4* rp = (const float4*)(&sc[col][0]);
#pragma unroll
        for (int q = 0; q < 8; ++q) {
            float4 rv = rp[q];
            vals[q * 4 + 0] = rv.x + endt[q * 4 + 0];
            vals[q * 4 + 1] = rv.y + endt[q * 4 + 1];
            vals[q * 4 + 2] = rv.z + endt[q * 4 + 2];
            vals[q * 4 + 3] = rv.w + endt[q * 4 + 3];
        }
    }
    float mx = vals[0];
#pragma unroll
    for (int j = 1; j < 32; ++j) mx = fmaxf(mx, vals[j]);
    float mxs = mx * LOG2E;
    float s = 0.f;
#pragma unroll
    for (int j = 0; j < 32; ++j) s += FAST_EXP2(__builtin_fmaf(vals[j], LOG2E, -mxs));
    if (hi == 0) fwd_out[b0 + col] = __builtin_fmaf(LN2, FAST_LOG2(s), mx);
}

// ---------------------------------------------------------------------------
// Gold path score: one thread per batch, plain gather-sum.
// ---------------------------------------------------------------------------
__global__ __launch_bounds__(256) void crf_gold(
    const float* __restrict__ em, const int* __restrict__ tags,
    const uint8_t* __restrict__ mask, const float* __restrict__ trans,
    const float* __restrict__ start, const float* __restrict__ endt,
    float* __restrict__ gold_out)
{
    int b = blockIdx.x * blockDim.x + threadIdx.x;
    if (b >= BB) return;
    const int*     tb = tags + (size_t)b * SS;
    const uint8_t* mb = mask + (size_t)b * SS;
    const float*   eb = em + (size_t)b * SS * LL;

    int   prev  = tb[0];
    float score = start[prev] + eb[prev];
    int   cnt   = mb[0] ? 1 : 0;
    for (int t = 1; t < SS; ++t) {
        int   tg = tb[t];
        float mf = mb[t] ? 1.f : 0.f;
        score += mf * (trans[prev * LL + tg] + eb[t * LL + tg]);
        cnt   += mb[t] ? 1 : 0;
        prev = tg;
    }
    int last_tag = tb[cnt - 1];
    gold_out[b] = score + endt[last_tag];
}

// ---------------------------------------------------------------------------
// Deterministic final reduction: mean(fwd - gold). Fixed-order tree, no atomics.
// ---------------------------------------------------------------------------
__global__ __launch_bounds__(256) void crf_reduce(
    const float* __restrict__ fwdv, const float* __restrict__ goldv,
    float* __restrict__ out)
{
    __shared__ float buf[256];
    int tid = threadIdx.x;
    float s = 0.f;
    for (int i = tid; i < BB; i += 256) s += fwdv[i] - goldv[i];
    buf[tid] = s;
    __syncthreads();
    for (int w = 128; w > 0; w >>= 1) {
        if (tid < w) buf[tid] += buf[tid + w];
        __syncthreads();
    }
    if (tid == 0) out[0] = buf[0] * (1.0f / BB);
}

extern "C" void kernel_launch(void* const* d_in, const int* in_sizes, int n_in,
                              void* d_out, int out_size, void* d_ws, size_t ws_size,
                              hipStream_t stream) {
    const float*   em    = (const float*)d_in[0];
    const int*     tags  = (const int*)d_in[1];
    const uint8_t* mask  = (const uint8_t*)d_in[2];
    const float*   trans = (const float*)d_in[3];
    const float*   start = (const float*)d_in[4];
    const float*   endt  = (const float*)d_in[5];
    float* out  = (float*)d_out;
    float* fwdv = (float*)d_ws;        // [BB]
    float* gldv = fwdv + BB;           // [BB]

    crf_forward<<<dim3(BB / TILE), dim3(32), 0, stream>>>(em, mask, trans, start,
                                                          endt, fwdv);
    crf_gold<<<dim3((BB + 255) / 256), dim3(256), 0, stream>>>(em, tags, mask, trans,
                                                               start, endt, gldv);
    crf_reduce<<<dim3(1), dim3(256), 0, stream>>>(fwdv, gldv, out);
}
